// HNetv1_38663295599042
// MI455X (gfx1250) — matrix operand order repaired and checked by reading, not compiled, not run on hardware
//
#include <hip/hip_runtime.h>

typedef __attribute__((ext_vector_type(16))) __bf16 v16bf;
typedef __attribute__((ext_vector_type(8)))  __bf16 v8bf;
typedef __attribute__((ext_vector_type(8)))  float  v8f;

#define NB   64      // batch
#define CCH  128     // channels
#define SPOS 144     // positions (12*12)
#define RIN  20736   // SPOS*SPOS

// ---------------------------------------------------------------------------
// K1: L2-normalize over channels and transpose to bf16 [n][pos][chan].
// One block per (batch, which-input). LDS-staged transpose, bank-padded.
// ---------------------------------------------------------------------------
__global__ __launch_bounds__(256) void norm_t_kernel(
    const float* __restrict__ x1, const float* __restrict__ x2,
    __bf16* __restrict__ o1, __bf16* __restrict__ o2) {
  const int n = blockIdx.x;
  const float* x = (blockIdx.y == 0) ? x1 : x2;
  __bf16* o = (blockIdx.y == 0) ? o1 : o2;

  __shared__ float rnorm[SPOS];
  __shared__ unsigned int buf[SPOS * 65];   // [pos][65 uints] = 130 halves, pad kills bank conflicts

  const int t = threadIdx.x;
  const float* base = x + (size_t)n * CCH * SPOS;

  // sum of squares over channels; consecutive threads -> consecutive pos (coalesced)
  if (t < SPOS) {
    float s = 0.f;
    #pragma unroll 4
    for (int c = 0; c < CCH; ++c) {
      float v = base[c * SPOS + t];
      s += v * v;
    }
    rnorm[t] = rsqrtf(s);
  }
  __syncthreads();

  // scale + transpose into LDS (coalesced global reads)
  for (int e = t; e < CCH * SPOS; e += 256) {
    int c = e / SPOS, p = e - c * SPOS;
    float v = base[e] * rnorm[p];
    ((__bf16*)buf)[p * 130 + c] = (__bf16)v;
  }
  __syncthreads();

  // coalesced dword copy-out, dropping the pad
  unsigned int* dst = (unsigned int*)(o + (size_t)n * SPOS * CCH);
  for (int idx = t; idx < SPOS * 64; idx += 256) {
    int p = idx >> 6, c2 = idx & 63;
    dst[idx] = buf[p * 65 + c2];
  }
}

// ---------------------------------------------------------------------------
// K2: per-batch correlation via v_wmma_f32_16x16x32_bf16.
// D[kpos, ij] = sum_c x2n[n][kpos][c] * x1n[n][ij][c]   (both stored [pos][chan])
// One wave per 16x16 output tile; K=128 in 4 steps of 32.
// Output written as bf16 into fc1 activation layout [n][kpos*144 + ij].
// ---------------------------------------------------------------------------
__global__ __launch_bounds__(256) void corr_kernel(
    const __bf16* __restrict__ x2t, const __bf16* __restrict__ x1t,
    __bf16* __restrict__ xact) {
  const int wid  = (int)((blockIdx.x * blockDim.x + threadIdx.x) >> 5);
  const int lane = (int)(threadIdx.x & 31);
  if (wid >= NB * 81) return;
  const int n  = wid / 81;
  const int tt = wid - n * 81;
  const int kt = tt / 9, jt = tt - kt * 9;

  const int l15 = lane & 15;
  const int kbA = (lane >> 4) * 16;   // A-fragment K base (contiguous, matches B)
  const int kbB = (lane >> 4) * 16;   // B-fragment K base

  const __bf16* arow = x2t + (size_t)(n * SPOS + kt * 16 + l15) * CCH + kbA;
  const __bf16* brow = x1t + (size_t)(n * SPOS + jt * 16 + l15) * CCH + kbB;

  v8f acc = {};
  #pragma unroll
  for (int c0 = 0; c0 < CCH; c0 += 32) {
    v8bf lo = *(const v8bf*)(arow + c0);
    v8bf hi = *(const v8bf*)(arow + c0 + 8);
    v16bf a;
    #pragma unroll
    for (int i = 0; i < 8; ++i) { a[i] = lo[i]; a[i + 8] = hi[i]; }
    v16bf b = *(const v16bf*)(brow + c0);
    acc = __builtin_amdgcn_wmma_f32_16x16x32_bf16(false, a, false, b,
                                                  (short)0, acc, false, false);
  }

  // D layout: VGPR r -> M = r + 8*(lane>>4), N = lane&15
  const int mhalf = (lane >> 4) * 8;
  __bf16* out = xact + (size_t)n * RIN + (size_t)(jt * 16 + l15);
  #pragma unroll
  for (int r = 0; r < 8; ++r)
    out[(size_t)(kt * 16 + r + mhalf) * SPOS] = (__bf16)acc[r];
}

// ---------------------------------------------------------------------------
// K3/K4/K5: activations[64xK](bf16) x weights[KxO](f32, cvt->bf16 in flight).
// One wave = one 16-col tile x all 64 rows (4 WMMA accumulators), K split
// KSPLIT ways into f32 partial buffers for occupancy. Weight bytes read
// exactly once, non-temporally (one-touch 430MB stream must not evict the
// heavily re-read activations from the 192MB L2).
// ---------------------------------------------------------------------------
template <int K, int O, int KSPLIT>
__global__ __launch_bounds__(128) void fc_gemm_kernel(
    const __bf16* __restrict__ act, const float* __restrict__ w,
    float* __restrict__ part) {
  constexpr int NT = (O + 15) / 16;
  constexpr int KSTEPS = (K + 31) / 32;
  constexpr int CS = (KSTEPS + KSPLIT - 1) / KSPLIT;
  constexpr bool OGUARD = (O % 16) != 0;   // fold guards away when shapes are tile-exact
  constexpr bool KGUARD = (K % 32) != 0;

  const int wid  = (int)((blockIdx.x * blockDim.x + threadIdx.x) >> 5);
  const int lane = (int)(threadIdx.x & 31);
  if (wid >= NT * KSPLIT) return;
  const int ot = wid % NT;
  const int ks = wid / NT;
  const int k_begin = ks * CS;
  const int k_end   = (KSTEPS < k_begin + CS) ? KSTEPS : (k_begin + CS);

  const int l15 = lane & 15;
  const int col = ot * 16 + l15;
  const bool colok = OGUARD ? (col < O) : true;
  const int kbA = (lane >> 4) * 8;
  const int kbB = (lane >> 4) * 16;
  const float* wc = w + col;

  v8f acc[4] = {v8f{}, v8f{}, v8f{}, v8f{}};

  for (int s = k_begin; s < k_end; ++s) {
    const int k0 = s * 32;

    // B fragment: f32 weights -> bf16, non-temporal (streamed once).
    // Lanes 0-15 cover K rows k0..k0+7,k0+16..k0+23; lanes 16-31 k0+8..k0+15,k0+24..k0+31;
    // 64B-coalesced across the 16 columns at each h.
    v16bf bfrag;
    #pragma unroll
    for (int h = 0; h < 16; ++h) {
      const int kk = k0 + kbA + (h & 7) + ((h >> 3) << 4);
      float wv = 0.f;
      if (colok && (!KGUARD || kk < K))
        wv = __builtin_nontemporal_load(wc + (size_t)kk * O);
      bfrag[h] = (__bf16)wv;
    }
    if (colok && s + 1 < k_end)
      __builtin_prefetch(wc + (size_t)(k0 + 32 + kbB) * O, 0, 1);

    #pragma unroll
    for (int mt = 0; mt < 4; ++mt) {
      const __bf16* rp = act + (size_t)(mt * 16 + l15) * K + k0 + kbA;
      v8bf lo = {}, hi = {};
      if (!KGUARD || k0 + kbA < K)      lo = *(const v8bf*)rp;
      if (!KGUARD || k0 + kbA + 16 < K) hi = *(const v8bf*)(rp + 16);
      v16bf a;
      #pragma unroll
      for (int i = 0; i < 8; ++i) { a[i] = lo[i]; a[i + 8] = hi[i]; }
      acc[mt] = __builtin_amdgcn_wmma_f32_16x16x32_bf16(false, a, false, bfrag,
                                                        (short)0, acc[mt], false, false);
    }
  }

  if (colok) {
    const int mhalf = (lane >> 4) * 8;
    #pragma unroll
    for (int mt = 0; mt < 4; ++mt)
      #pragma unroll
      for (int r = 0; r < 8; ++r)
        part[((size_t)ks * 64 + mt * 16 + r + mhalf) * O + col] = acc[mt][r];
  }
}

// ---------------------------------------------------------------------------
// Reduce KSPLIT partials + bias + activation; emit bf16 and/or f32.
// actType: 0 = ReLU, 1 = tanh.
// ---------------------------------------------------------------------------
__global__ __launch_bounds__(256) void reduce_act_kernel(
    const float* __restrict__ part, const float* __restrict__ bias,
    int O, int ksplit, int actType,
    __bf16* __restrict__ obf, float* __restrict__ of32) {
  const int total = NB * O;
  const int idx = (int)(blockIdx.x * blockDim.x + threadIdx.x);
  if (idx >= total) return;
  const int o = idx % O;
  float s = 0.f;
  for (int k = 0; k < ksplit; ++k) s += part[(size_t)k * total + idx];
  s += bias[o];
  s = (actType == 0) ? fmaxf(s, 0.f) : tanhf(s);
  if (obf)  obf[idx]  = (__bf16)s;
  if (of32) of32[idx] = s;
}

// ---------------------------------------------------------------------------
// K6: fc4 (324 -> 8) + append 1.0, write [64,3,3] f32.
// ---------------------------------------------------------------------------
__global__ __launch_bounds__(256) void head_kernel(
    const float* __restrict__ act3, const float* __restrict__ w4,
    const float* __restrict__ b4, float* __restrict__ out) {
  const int t = (int)(blockIdx.x * blockDim.x + threadIdx.x);
  if (t >= NB * 8) return;
  const int n = t >> 3, o = t & 7;
  float s = b4[o];
  for (int k = 0; k < 324; ++k) s += act3[n * 324 + k] * w4[k * 8 + o];
  out[n * 9 + o] = s;
  if (o == 0) out[n * 9 + 8] = 1.0f;
}

// ---------------------------------------------------------------------------
extern "C" void kernel_launch(void* const* d_in, const int* in_sizes, int n_in,
                              void* d_out, int out_size, void* d_ws, size_t ws_size,
                              hipStream_t stream) {
  const float* x1 = (const float*)d_in[0];
  const float* x2 = (const float*)d_in[1];
  const float* w1 = (const float*)d_in[2];
  const float* b1 = (const float*)d_in[3];
  const float* w2 = (const float*)d_in[4];
  const float* b2 = (const float*)d_in[5];
  const float* w3 = (const float*)d_in[6];
  const float* b3 = (const float*)d_in[7];
  const float* w4 = (const float*)d_in[8];
  const float* b4 = (const float*)d_in[9];
  float* out = (float*)d_out;

  char* ws = (char*)d_ws;
  size_t off = 0;
  auto alloc = [&](size_t bytes) -> char* {
    char* p = ws + off;
    off += (bytes + 255) & ~(size_t)255;
    return p;
  };
  __bf16* x1t   = (__bf16*)alloc((size_t)NB * SPOS * CCH * 2);
  __bf16* x2t   = (__bf16*)alloc((size_t)NB * SPOS * CCH * 2);
  __bf16* xact  = (__bf16*)alloc((size_t)NB * RIN * 2);
  float*  part1 = (float*) alloc((size_t)16 * NB * 5184 * 4);
  __bf16* act1  = (__bf16*)alloc((size_t)NB * 5184 * 2);
  float*  part2 = (float*) alloc((size_t)16 * NB * 1296 * 4);
  __bf16* act2  = (__bf16*)alloc((size_t)NB * 1296 * 2);
  float*  part3 = (float*) alloc((size_t)32 * NB * 324 * 4);
  float*  act3f = (float*) alloc((size_t)NB * 324 * 4);

  // K1: normalize + transpose to bf16
  norm_t_kernel<<<dim3(NB, 2), 256, 0, stream>>>(x1, x2, x1t, x2t);

  // K2: correlation GEMM (64*81 waves, 8 waves/block)
  corr_kernel<<<(NB * 81) / 8, 256, 0, stream>>>(x2t, x1t, xact);

  // fc1: 20736 -> 5184, KSPLIT=16 -> 324*16 = 5184 waves
  {
    int waves = (5184 / 16) * 16;
    fc_gemm_kernel<20736, 5184, 16><<<(waves + 3) / 4, 128, 0, stream>>>(xact, w1, part1);
    int total = NB * 5184;
    reduce_act_kernel<<<(total + 255) / 256, 256, 0, stream>>>(part1, b1, 5184, 16, 0, act1, nullptr);
  }
  // fc2: 5184 -> 1296, KSPLIT=16 -> 81*16 = 1296 waves
  {
    int waves = (1296 / 16) * 16;
    fc_gemm_kernel<5184, 1296, 16><<<(waves + 3) / 4, 128, 0, stream>>>(act1, w2, part2);
    int total = NB * 1296;
    reduce_act_kernel<<<(total + 255) / 256, 256, 0, stream>>>(part2, b2, 1296, 16, 0, act2, nullptr);
  }
  // fc3: 1296 -> 324 (tanh, keep f32 output for the tiny head), KSPLIT=32
  {
    int waves = ((324 + 15) / 16) * 32;
    fc_gemm_kernel<1296, 324, 32><<<(waves + 3) / 4, 128, 0, stream>>>(act2, w3, part3);
    int total = NB * 324;
    reduce_act_kernel<<<(total + 255) / 256, 256, 0, stream>>>(part3, b3, 324, 32, 1, nullptr, act3f);
  }
  // fc4 + ones
  head_kernel<<<2, 256, 0, stream>>>(act3f, w4, b4, out);
}
